// ReservoirDecoder_85521388798432
// MI455X (gfx1250) — matrix operand order repaired and checked by reading, not compiled, not run on hardware
//
#include <hip/hip_runtime.h>
#include <stdint.h>

// ---------------------------------------------------------------------------
// Types
// ---------------------------------------------------------------------------
typedef __attribute__((ext_vector_type(16))) _Float16 v16h;
typedef __attribute__((ext_vector_type(8)))  _Float16 v8h;
typedef __attribute__((ext_vector_type(8)))  float    v8f;
typedef __attribute__((ext_vector_type(4)))  int      v4i;

#define BB 8
#define TT 16
#define ZD 128
#define HC 64
#define HH 64
#define WW 64
#define NGATES 256   // 4*HC

#define AS_GLOBAL __attribute__((address_space(1)))
#define AS_LOCAL  __attribute__((address_space(3)))

// Async global->LDS path (gfx1250). Guarded: falls back to synchronous
// copies if the builtins are absent in this toolchain.
#if defined(__AMDGCN__) && __has_builtin(__builtin_amdgcn_global_load_async_to_lds_b128) && __has_builtin(__builtin_amdgcn_s_wait_asynccnt)
#define USE_ASYNC_LDS 1
#else
#define USE_ASYNC_LDS 0
#endif

// ---------------------------------------------------------------------------
// Device math helpers
// ---------------------------------------------------------------------------
__device__ __forceinline__ float sigmoid_f(float x) {
    return 1.0f / (1.0f + __expf(-x));
}
__device__ __forceinline__ float tanh_f(float x) {
    float e2 = __expf(-2.0f * fabsf(x));
    float t  = (1.0f - e2) / (1.0f + e2);
    return copysignf(t, x);
}

// ---------------------------------------------------------------------------
// Linear projection:  x = relu(z @ lin_w^T + b)  -> xall (T,B,4096) f16
// z: (B,T,128) f32, lin_w: (4096,128) f32
// ---------------------------------------------------------------------------
__global__ void linear_relu_kernel(const float* __restrict__ z,
                                   const float* __restrict__ w,
                                   const float* __restrict__ b,
                                   _Float16* __restrict__ xall) {
    int gid = blockIdx.x * blockDim.x + threadIdx.x;   // 524288
    int o   = gid & 4095;
    int bt  = gid >> 12;            // b*T + t
    int bb  = bt >> 4;
    int t   = bt & 15;
    const float* zr = z + (size_t)bt * ZD;
    const float* wr = w + (size_t)o * ZD;
    float acc = b[o];
#pragma unroll 8
    for (int k = 0; k < ZD; ++k) acc = fmaf(zr[k], wr[k], acc);
    acc = fmaxf(acc, 0.0f);
    xall[((size_t)t * BB + bb) * 4096 + o] = (_Float16)acc;
}

// ---------------------------------------------------------------------------
// Weight packing into the WMMA dense-B fragment layout.
// Dense 16-bit B 32x16: lane holds column n = lane&15;
// element e -> k = 8h+e (e<8), 16+8h+(e-8) (e>=8), h = lane>>4 (matches A).
// Packed layout: dst[((chunk*16 + nt)*32 + lane)*16 + e], chunk = tap*KC + kc.
// layer0 input channel map (Cpad=96): ch<64 -> h (cin=1+ch), ch==64 -> x (cin=0), else 0
// layer1 (Cpad=128): ch -> cin directly.
// ---------------------------------------------------------------------------
__global__ void pack_w_kernel(const float* __restrict__ wsrc,
                              _Float16* __restrict__ dst,
                              int KC, int nchunks, int layer) {
    int gid = blockIdx.x * blockDim.x + threadIdx.x;
    int tot = nchunks * 16 * 32 * 16;
    if (gid >= tot) return;
    int e     = gid & 15;
    int lane  = (gid >> 4) & 31;
    int nt    = (gid >> 9) & 15;
    int chunk = gid >> 13;
    int tap = chunk / KC, kc = chunk - tap * KC;
    int ky = tap / 3, kx = tap - ky * 3;
    int kB = ((lane & 16) >> 1) + (e & 7) + ((e & 8) << 1);  // B-matrix k mapping
    int ch = kc * 32 + kB;
    int n  = nt * 16 + (lane & 15);
    float val = 0.0f;
    if (layer == 0) {
        if (ch < 64)       val = wsrc[((n * 65 + (1 + ch)) * 3 + ky) * 3 + kx];
        else if (ch == 64) val = wsrc[((n * 65 + 0) * 3 + ky) * 3 + kx];
    } else {
        if (ch < 128)      val = wsrc[((n * 128 + ch) * 3 + ky) * 3 + kx];
    }
    dst[gid] = (_Float16)val;
}

// ---------------------------------------------------------------------------
// Copy frame t into layer0 input buffer, channel 64.
// ---------------------------------------------------------------------------
__global__ void pack_x_kernel(const _Float16* __restrict__ xall,
                              _Float16* __restrict__ X0, int t) {
    int gid = blockIdx.x * blockDim.x + threadIdx.x;  // 32768 = B*H*W
    X0[(size_t)gid * 96 + 64] = xall[(size_t)t * 32768 + gid];
}

// ---------------------------------------------------------------------------
// Fused ConvLSTM step: implicit-GEMM 3x3 conv via WMMA + gate nonlinearity.
//  Block = 128 threads (4 waves) = one image row (b,y): 64 pixels.
//  Wave w owns M-tile pixels x0=16w..16w+15 and all 16 N-tiles (256 gates).
//  K loop: 9 taps x KC chunks of 32 channels, channels-last input.
//  B chunks staged into double-buffered LDS with async global->LDS.
// ---------------------------------------------------------------------------
__global__ __launch_bounds__(128)
void convlstm_step_kernel(const _Float16* __restrict__ Xin, int Cpad, int KC,
                          const _Float16* __restrict__ Wpk,
                          const float* __restrict__ bias,
                          const float* __restrict__ Cprev,
                          float* __restrict__ Cout,
                          _Float16* __restrict__ Hd1, int Cd1, int off1,
                          _Float16* __restrict__ Hd2, int Cd2, int off2,
                          int relu2) {
    __shared__ __attribute__((aligned(32))) _Float16 sB[2][8192];  // 2 x 16KB

    const int lane = threadIdx.x & 31;
    const int wave = threadIdx.x >> 5;
    const int row  = blockIdx.x;          // b*64 + y
    const int b    = row >> 6;
    const int y    = row & 63;
    const int x0   = wave * 16;
    const int nchunks = 9 * KC;

    // Cooperative B-chunk loader: 16KB per chunk, 128 threads x 8 x 16B.
    auto loadB = [&](int chunk, int bufi) {
        const _Float16* src = Wpk + (size_t)chunk * 8192;
#if USE_ASYNC_LDS
#pragma unroll
        for (int i = 0; i < 8; ++i) {
            int off = (i * 128 + (int)threadIdx.x) * 8;   // halves (16B granule)
            __builtin_amdgcn_global_load_async_to_lds_b128(
                (AS_GLOBAL v4i*)(src + off),
                (AS_LOCAL v4i*)(&sB[bufi][off]),
                0, 0);
        }
#else
#pragma unroll
        for (int i = 0; i < 8; ++i) {
            int off = (i * 128 + (int)threadIdx.x) * 8;
            *(v8h*)(&sB[bufi][off]) = *(const v8h*)(src + off);
        }
#endif
    };

    // Accumulators: 16 N-tiles of 16x16 f32 (8 VGPRs each), bias pre-splatted.
    v8f acc[16];
#pragma unroll
    for (int nt = 0; nt < 16; ++nt) {
        float bv = bias[nt * 16 + (lane & 15)];
        v8f a8 = {bv, bv, bv, bv, bv, bv, bv, bv};
        acc[nt] = a8;
    }

    const int klo = (lane & 16) >> 1;    // A-fragment: lanes>=16 start at K=8
    const int m   = lane & 15;           // A-fragment row (pixel within tile)

    int buf = 0;
    int chunk = 0;
    loadB(0, 0);

    for (int tap = 0; tap < 9; ++tap) {
        const int ky = tap / 3;
        const int kx = tap - ky * 3;
        const int sy = y + ky - 1;
        const int sx = x0 + m + kx - 1;
        const bool valid = ((unsigned)sy < (unsigned)HH) & ((unsigned)sx < (unsigned)WW);
        const int pidx = valid ? ((b << 12) + (sy << 6) + sx) : 0;
        const _Float16* ap = Xin + (size_t)pidx * Cpad + klo;

        for (int kc = 0; kc < KC; ++kc, ++chunk) {
            // Wait for this wave's async fills of sB[buf], then converge.
#if USE_ASYNC_LDS
            __builtin_amdgcn_s_wait_asynccnt(0);
#endif
            __syncthreads();
            // Prefetch next chunk into the other buffer (overlaps compute).
            if (chunk + 1 < nchunks) loadB(chunk + 1, buf ^ 1);

            // A fragment (16-bit A 16x32 layout): two 16B runs per lane.
            v8h lo = {}; v8h hi = {};
            if (valid) {
                lo = *(const v8h*)(ap + kc * 32);
                hi = *(const v8h*)(ap + kc * 32 + 16);
            }
            v16h afrag = __builtin_shufflevector(lo, hi,
                0, 1, 2, 3, 4, 5, 6, 7, 8, 9, 10, 11, 12, 13, 14, 15);

#pragma unroll
            for (int nt = 0; nt < 16; ++nt) {
                v16h bfrag = *(const v16h*)(&sB[buf][(nt * 32 + lane) * 16]);
                acc[nt] = __builtin_amdgcn_wmma_f32_16x16x32_f16(
                    false, afrag, false, bfrag, (short)0, acc[nt], false, false);
            }
            buf ^= 1;
        }
    }

    // Epilogue: LSTM gates. C/D layout: lane -> col n=lane&15, vgpr v -> row m
    // (m += 8 for lanes 16-31). Gate blocks: i=n, f=64+n, o=128+n, g=192+n.
    const int mhi = (lane & 16) >> 1;
    const int cnb = lane & 15;
    const size_t pixrow = (size_t)((b << 12) + (y << 6) + x0);
#pragma unroll
    for (int v = 0; v < 8; ++v) {
        const size_t px = pixrow + v + mhi;
#pragma unroll
        for (int q = 0; q < 4; ++q) {
            const int cn = q * 16 + cnb;
            float iv = acc[q][v];
            float fv = acc[q + 4][v];
            float ov = acc[q + 8][v];
            float gv = acc[q + 12][v];
            float cp = Cprev[px * HC + cn];
            float cnew = sigmoid_f(fv) * cp + sigmoid_f(iv) * tanh_f(gv);
            float hnew = sigmoid_f(ov) * tanh_f(cnew);
            Cout[px * HC + cn] = cnew;
            Hd1[px * Cd1 + off1 + cn] = (_Float16)hnew;
            float h2 = relu2 ? fmaxf(hnew, 0.0f) : hnew;
            Hd2[px * Cd2 + off2 + cn] = (_Float16)h2;
        }
    }
}

// ---------------------------------------------------------------------------
// Decoder: out = sigmoid(Conv3d(relu_h1, dec_w, pad=1) + dec_b)
// ys: (T,B,H,W,64) f16 (already relu'd). out: (B,T,1,H,W) f32.
// ys is L2-resident (67MB < 192MB); VALU is plenty for 1.8 GFLOP.
// ---------------------------------------------------------------------------
__global__ void decoder_kernel(const _Float16* __restrict__ ys,
                               const float* __restrict__ dec_w,
                               const float* __restrict__ dec_b,
                               float* __restrict__ out) {
    __shared__ float wds[27 * 64];
    for (int i = threadIdx.x; i < 27 * 64; i += blockDim.x) {
        int tap = i >> 6, c = i & 63;
        int dt = tap / 9, r = tap - dt * 9;
        int dy = r / 3, dx = r - dy * 3;
        // dec_w shape (1, 64, 3, 3, 3): [c][dt][dy][dx]
        wds[i] = dec_w[((c * 3 + dt) * 3 + dy) * 3 + dx];
    }
    __syncthreads();

    int gid = blockIdx.x * blockDim.x + threadIdx.x;  // 524288 outputs
    int px = gid & 4095;
    int bt = gid >> 12;
    int t  = bt & 15;
    int bb = bt >> 4;
    int yy = px >> 6, xx = px & 63;

    float acc = dec_b[0];
    for (int dt = 0; dt < 3; ++dt) {
        int st = t + dt - 1;
        if ((unsigned)st >= (unsigned)TT) continue;
        for (int dy = 0; dy < 3; ++dy) {
            int sy = yy + dy - 1;
            if ((unsigned)sy >= (unsigned)HH) continue;
            for (int dx = 0; dx < 3; ++dx) {
                int sx = xx + dx - 1;
                if ((unsigned)sx >= (unsigned)WW) continue;
                const v8h* hp = (const v8h*)(ys +
                    (((size_t)st * BB + bb) * 4096 + (size_t)sy * 64 + sx) * HC);
                const float* wp = &wds[((dt * 3 + dy) * 3 + dx) * 64];
#pragma unroll
                for (int c8 = 0; c8 < 8; ++c8) {
                    v8h hv = hp[c8];
#pragma unroll
                    for (int j = 0; j < 8; ++j)
                        acc = fmaf((float)hv[j], wp[c8 * 8 + j], acc);
                }
            }
        }
    }
    out[gid] = 1.0f / (1.0f + __expf(-acc));
}

// ---------------------------------------------------------------------------
// Host-side orchestration (graph-capture safe: kernels + memsetAsync only)
// ---------------------------------------------------------------------------
extern "C" void kernel_launch(void* const* d_in, const int* in_sizes, int n_in,
                              void* d_out, int out_size, void* d_ws, size_t ws_size,
                              hipStream_t stream) {
    (void)in_sizes; (void)n_in; (void)out_size; (void)ws_size;

    const float* z     = (const float*)d_in[0];   // (B,T,128)
    const float* lin_w = (const float*)d_in[1];   // (4096,128)
    const float* lin_b = (const float*)d_in[2];   // (4096)
    const float* cw0   = (const float*)d_in[3];   // (256,65,3,3)
    const float* cb0   = (const float*)d_in[4];   // (256)
    const float* cw1   = (const float*)d_in[5];   // (256,128,3,3)
    const float* cb1   = (const float*)d_in[6];   // (256)
    const float* dw    = (const float*)d_in[7];   // (1,64,3,3,3)
    const float* db    = (const float*)d_in[8];   // (1)
    float* out = (float*)d_out;

    // Workspace carve-up (256B aligned)
    char* ws = (char*)d_ws;
    size_t off = 0;
    auto carve = [&](size_t bytes) -> char* {
        char* p = ws + off;
        off = (off + bytes + 255) & ~(size_t)255;
        return p;
    };
    _Float16* WPK0  = (_Float16*)carve((size_t)27 * 16384);       // 432 KB
    _Float16* WPK1  = (_Float16*)carve((size_t)36 * 16384);       // 576 KB
    _Float16* XALL  = (_Float16*)carve((size_t)524288 * 2);       // 1 MB
    _Float16* X0[2] = {(_Float16*)carve((size_t)8 * 4096 * 96 * 2),
                       (_Float16*)carve((size_t)8 * 4096 * 96 * 2)};
    _Float16* X1[2] = {(_Float16*)carve((size_t)8 * 4096 * 128 * 2),
                       (_Float16*)carve((size_t)8 * 4096 * 128 * 2)};
    float* C0 = (float*)carve((size_t)8 * 4096 * 64 * 4);
    float* C1 = (float*)carve((size_t)8 * 4096 * 64 * 4);
    _Float16* YS = (_Float16*)carve((size_t)16 * 8 * 4096 * 64 * 2);  // 67 MB

    // Zero-init recurrent state + padded channels (deterministic each call).
    (void)hipMemsetAsync(X0[0], 0, (size_t)8 * 4096 * 96 * 2, stream);
    (void)hipMemsetAsync(X0[1], 0, (size_t)8 * 4096 * 96 * 2, stream);
    (void)hipMemsetAsync(X1[0], 0, (size_t)8 * 4096 * 128 * 2, stream);
    (void)hipMemsetAsync(X1[1], 0, (size_t)8 * 4096 * 128 * 2, stream);
    (void)hipMemsetAsync(C0, 0, (size_t)8 * 4096 * 64 * 4, stream);
    (void)hipMemsetAsync(C1, 0, (size_t)8 * 4096 * 64 * 4, stream);

    // Pack weights into WMMA B-fragment order.
    pack_w_kernel<<<(27 * 8192 + 255) / 256, 256, 0, stream>>>(cw0, WPK0, 3, 27, 0);
    pack_w_kernel<<<(36 * 8192 + 255) / 256, 256, 0, stream>>>(cw1, WPK1, 4, 36, 1);

    // Linear + ReLU -> frames.
    linear_relu_kernel<<<524288 / 256, 256, 0, stream>>>(z, lin_w, lin_b, XALL);

    // Recurrence: 16 timesteps, two fused WMMA ConvLSTM layers per step.
    for (int t = 0; t < TT; ++t) {
        int pi = t & 1, po = pi ^ 1;
        pack_x_kernel<<<32768 / 256, 256, 0, stream>>>(XALL, X0[pi], t);

        // Layer 0: gates = conv3x3([h0|x]) ; h0_new -> X1[pi] ch0..63 (this
        // step's layer-1 input) and X0[po] ch0..63 (next step's layer-0 input).
        convlstm_step_kernel<<<512, 128, 0, stream>>>(
            X0[pi], 96, 3, WPK0, cb0, C0, C0,
            X1[pi], 128, 0,
            X0[po], 96, 0, 0);

        // Layer 1: gates = conv3x3([h0_new|h1]) ; h1_new -> X1[po] ch64..127
        // (next step) and relu(h1_new) -> ys[t] (decoder staging).
        convlstm_step_kernel<<<512, 128, 0, stream>>>(
            X1[pi], 128, 4, WPK1, cb1, C1, C1,
            X1[po], 128, 64,
            YS + (size_t)t * 8 * 4096 * 64, 64, 0, 1);
    }

    // Decoder Conv3d + sigmoid.
    decoder_kernel<<<524288 / 256, 256, 0, stream>>>(YS, dw, db, out);
}
